// AggregateOrInclusive_16535624090066
// MI455X (gfx1250) — matrix operand order repaired and checked by reading, not compiled, not run on hardware
//
#include <hip/hip_runtime.h>
#include <hip/hip_bf16.h>
#include <stdint.h>

// Problem constants (from reference)
constexpr int kS    = 8;
constexpr int kIN   = 256;
constexpr int kHID  = 1024;
constexpr int kH    = 8;
constexpr int kAE   = 64;
constexpr int kOE   = 128;
constexpr int kB    = 8192;
constexpr int kHAE  = kH * kAE;   // 512
constexpr int kHOE  = kH * kOE;   // 1024
constexpr int kXROW = kS * kIN + kS;  // 2056 floats per input row
constexpr int kOROW = kHOE + kS;      // 1032 floats per output row

// TDM-staged LDS tile geometry: 16 rows x 2048B, 8B pad after each 1024B chunk
constexpr int kRowB = 2064;  // LDS row stride in bytes (2048 data + 2x8 pad)

// CDNA5 WMMA / TDM types
typedef __attribute__((ext_vector_type(16))) __bf16   v16bf;
typedef __attribute__((ext_vector_type(8)))  float    v8f;
typedef __attribute__((ext_vector_type(4)))  unsigned v4u;
typedef __attribute__((ext_vector_type(2)))  unsigned v2u;
typedef __attribute__((ext_vector_type(4)))  float    v4f;
typedef __attribute__((ext_vector_type(8)))  int      v8i;
typedef __attribute__((ext_vector_type(4)))  int      v4i;

union BF16x16 { v16bf v; v4u u[2]; v2u u2[4]; };  // 32B WMMA bf16 operand
union BF16x4  { __bf16 h[4]; v2u u; };            // 8B pack

static __device__ __forceinline__ v8f wmma_bf16(v16bf a, v16bf b, v8f c) {
  return __builtin_amdgcn_wmma_f32_16x16x32_bf16(false, a, false, b, (short)0, c,
                                                 false, false);
}

// ---------------------------------------------------------------------------
// TDM: DMA a 16-row x 2048B tile (row stride 16KB) of bf16 xact into LDS,
// inserting 8B of padding after every 1024B for bank-conflict-free reads.
// Issue from ONE wave only; follow with s_wait_tensorcnt + workgroup barrier.
// (6-arg builtin form: g0, g1, g2, g3, g4, cpol)
// ---------------------------------------------------------------------------
static __device__ __forceinline__ void tdm_load_xact_tile(
    const __bf16* gsrc, unsigned lds_byte_addr) {
  unsigned long long ga = (unsigned long long)(uintptr_t)gsrc;
  v4u g0;
  g0[0] = 1u;                                         // count=1, user desc
  g0[1] = lds_byte_addr;                              // lds_addr
  g0[2] = (unsigned)(ga & 0xFFFFFFFFu);               // global_addr lo
  g0[3] = (unsigned)((ga >> 32) & 0x1FFFFFFu) | (2u << 30);  // addr hi | type=2
  v8i g1;
  // data_size=3 (8B) | pad_enable | pad_interval=7 (256 dw = 1024B) |
  // pad_amount=1 (2 dw = 8B)
  g1[0] = (int)((3u << 16) | (1u << 20) | (7u << 22) | (1u << 25));
  g1[1] = (int)(256u << 16);      // tensor_dim0 = 256 (8B units)
  g1[2] = (int)(16u << 16);       // tensor_dim1 = 16
  g1[3] = (int)(256u << 16);      // tile_dim0 = 256
  g1[4] = 16;                     // tile_dim1 = 16, tile_dim2 = 0
  g1[5] = 2048;                   // tensor_dim0_stride = 16KB / 8B
  g1[6] = 0;
  g1[7] = 0;
  v4i z4 = {0, 0, 0, 0};
  v8i z8 = {0, 0, 0, 0, 0, 0, 0, 0};
  __builtin_amdgcn_tensor_load_to_lds(g0, g1, z4, z4, z8, 0);
}

// A-operand fetch from a TDM-padded LDS row: element k -> byte 2k + 8*(k/512)
static __device__ __forceinline__ v2u lds_a8(const char* rowb, int k) {
  return *(const v2u*)(rowb + 2 * k + ((k >> 9) << 3));
}
static __device__ __forceinline__ BF16x16 lds_a_operand(const char* rowb, int ks,
                                                        int hb) {
  const int k1 = ks * 32 + hb * 16;
  const int k2 = k1 + 8;
  BF16x16 A;
  A.u2[0] = lds_a8(rowb, k1);
  A.u2[1] = lds_a8(rowb, k1 + 4);
  A.u2[2] = lds_a8(rowb, k2);
  A.u2[3] = lds_a8(rowb, k2 + 4);
  return A;
}

// ---------------------------------------------------------------------------
// Kernel 0: transpose + convert weights f32 [s][k][n] -> bf16 [s][n][k]
// ---------------------------------------------------------------------------
__global__ __launch_bounds__(256) void transpose_bf16_kernel(
    const float* __restrict__ src, __bf16* __restrict__ dst,
    int K, int N, long total) {
  long i = (long)blockIdx.x * 256 + threadIdx.x;
  if (i >= total) return;
  int  k = (int)(i % K);
  long r = i / K;
  int  n = (int)(r % N);
  long s = r / N;
  dst[i] = (__bf16)src[(s * (long)K + k) * (long)N + n];
}

// ---------------------------------------------------------------------------
// Kernel 1: xact[b,s,:] = relu(feat[b,s,:] @ W_sub[s] + b_sub[s]) -> bf16
// ---------------------------------------------------------------------------
__global__ __launch_bounds__(256) void sub_gemm_kernel(
    const float* __restrict__ x, const __bf16* __restrict__ WtSub,
    const float* __restrict__ bSub, __bf16* __restrict__ Xact) {
  const int s  = blockIdx.x;
  const int b0 = blockIdx.y * 64;
  __shared__ __bf16 featLds[64 * 264];  // K=256 + 8 pad, conflict-free

  const int t = threadIdx.x;
  #pragma unroll
  for (int j = 0; j < 16; ++j) {
    int c = t + j * 256;
    int row = c >> 6, fi = c & 63;
    v4f f = *(const v4f*)(x + (size_t)(b0 + row) * kXROW + s * kIN + fi * 4);
    BF16x4 p;
    p.h[0] = (__bf16)f[0]; p.h[1] = (__bf16)f[1];
    p.h[2] = (__bf16)f[2]; p.h[3] = (__bf16)f[3];
    *(v2u*)&featLds[row * 264 + fi * 4] = p.u;
  }
  __syncthreads();

  const int lane = t & 31, wave = t >> 5;
  const int l15 = lane & 15, hb = lane >> 4;
  const int mt = wave & 3;
  const int nh = wave >> 2;

  BF16x16 A[8];
  const __bf16* ap = &featLds[(mt * 16 + l15) * 264];
  #pragma unroll
  for (int ks = 0; ks < 8; ++ks) {
    A[ks].u[0] = *(const v4u*)(ap + ks * 32 + hb * 16);
    A[ks].u[1] = *(const v4u*)(ap + ks * 32 + hb * 16 + 8);
  }

  for (int nt = nh * 32; nt < nh * 32 + 32; ++nt) {
    const int n = nt * 16 + l15;
    const __bf16* bp = WtSub + ((size_t)s * kHID + n) * kIN + hb * 16;
    v8f acc = {};
    #pragma unroll
    for (int ks = 0; ks < 8; ++ks) {
      BF16x16 Bv;
      Bv.u[0] = *(const v4u*)(bp + ks * 32);
      Bv.u[1] = *(const v4u*)(bp + ks * 32 + 8);
      acc = wmma_bf16(A[ks].v, Bv.v, acc);
    }
    const float bias = bSub[s * kHID + n];
    #pragma unroll
    for (int i = 0; i < 8; ++i) {
      int row = i + hb * 8;
      float val = fmaxf(acc[i] + bias, 0.f);
      Xact[((size_t)(b0 + mt * 16 + row) * kS + s) * kHID + n] = (__bf16)val;
    }
  }
}

// ---------------------------------------------------------------------------
// Kernel 2: fused q/k GEMMs -> scores[b,s,h]; xact tile staged via TDM
// ---------------------------------------------------------------------------
__global__ __launch_bounds__(256) void qk_score_kernel(
    const __bf16* __restrict__ Xact,
    const __bf16* __restrict__ Wqt, const float* __restrict__ bq,
    const __bf16* __restrict__ Wkt, const float* __restrict__ bk,
    float* __restrict__ Scores) {
  const int s  = blockIdx.x;
  const int b0 = blockIdx.y * 16;
  __shared__ __bf16 xLds[16 * (kRowB / 2)];  // 16 x 2064B (TDM-padded)

  const int t = threadIdx.x;
  if (t < 32) {  // one wave issues the DMA
    tdm_load_xact_tile(Xact + ((size_t)b0 * kS + s) * kHID,
                       (unsigned)(uintptr_t)&xLds[0]);
    __builtin_amdgcn_s_wait_tensorcnt(0);
  }
  __syncthreads();

  const int lane = t & 31, w = t >> 5;  // w = head
  const int l15 = lane & 15, hb = lane >> 4;
  const char* rowb = (const char*)&xLds[0] + l15 * kRowB;

  float p[8] = {0, 0, 0, 0, 0, 0, 0, 0};
  for (int nt = 0; nt < 4; ++nt) {
    const int n = w * 64 + nt * 16 + l15;
    const __bf16* bqp = Wqt + ((size_t)s * kHAE + n) * kHID + hb * 16;
    const __bf16* bkp = Wkt + ((size_t)s * kHAE + n) * kHID + hb * 16;
    v8f accq = {}, acck = {};
    #pragma unroll
    for (int ks = 0; ks < 32; ++ks) {
      BF16x16 Av = lds_a_operand(rowb, ks, hb);
      BF16x16 Bq, Bk;
      Bq.u[0] = *(const v4u*)(bqp + ks * 32);
      Bq.u[1] = *(const v4u*)(bqp + ks * 32 + 8);
      Bk.u[0] = *(const v4u*)(bkp + ks * 32);
      Bk.u[1] = *(const v4u*)(bkp + ks * 32 + 8);
      accq = wmma_bf16(Av.v, Bq.v, accq);
      acck = wmma_bf16(Av.v, Bk.v, acck);
    }
    const float qb = bq[s * kHAE + n], kb = bk[s * kHAE + n];
    #pragma unroll
    for (int i = 0; i < 8; ++i) p[i] += (accq[i] + qb) * (acck[i] + kb);
  }
  #pragma unroll
  for (int i = 0; i < 8; ++i) {
    float v = p[i];
    for (int m = 1; m < 16; m <<= 1) v += __shfl_xor(v, m, 32);
    p[i] = v;
  }
  if (l15 == 0) {
    #pragma unroll
    for (int i = 0; i < 8; ++i) {
      int row = i + hb * 8;
      Scores[((size_t)(b0 + row) * kS + s) * kH + w] = p[i] * 0.125f;
    }
  }
}

// ---------------------------------------------------------------------------
// Kernel 3: masked softmax over S (+1 denom) -> att; mask passthrough
// ---------------------------------------------------------------------------
__global__ __launch_bounds__(256) void att_kernel(
    const float* __restrict__ x, const float* __restrict__ Scores,
    float* __restrict__ Att, float* __restrict__ out) {
  const int t = blockIdx.x * 256 + threadIdx.x;
  const int b = t >> 3, h = t & 7;
  float m[kS], z[kS], ez[kS];
  float zmax = 0.f;
  #pragma unroll
  for (int s = 0; s < kS; ++s) {
    m[s] = x[(size_t)b * kXROW + kS * kIN + s];
    float sc = Scores[((size_t)b * kS + s) * kH + h];
    z[s] = m[s] * sc;
    zmax = fmaxf(zmax, fmaxf(z[s] * m[s], 0.f));
  }
  float denom = 1.f;
  #pragma unroll
  for (int s = 0; s < kS; ++s) { ez[s] = __expf(z[s] - zmax); denom += m[s] * ez[s]; }
  #pragma unroll
  for (int s = 0; s < kS; ++s)
    Att[((size_t)b * kS + s) * kH + h] = m[s] * ez[s] / denom;
  if (h == 0) {
    #pragma unroll
    for (int s = 0; s < kS; ++s)
      out[(size_t)b * kOROW + kHOE + s] = m[s];
  }
}

// ---------------------------------------------------------------------------
// Kernel 4: v-GEMM fused with attention pooling; xact tiles staged via TDM
// ---------------------------------------------------------------------------
__global__ __launch_bounds__(256) void pool_kernel(
    const __bf16* __restrict__ Xact, const __bf16* __restrict__ Wvt,
    const float* __restrict__ bv, const float* __restrict__ Att,
    float* __restrict__ out) {
  const int b0 = blockIdx.x * 16;
  __shared__ __bf16 xLds[16 * (kRowB / 2)];
  __shared__ float  attLds[16 * 64];  // [row][s*8+h]

  const int t = threadIdx.x;
  #pragma unroll
  for (int j = 0; j < 4; ++j) {
    int c = t + j * 256;
    attLds[c] = Att[(size_t)b0 * 64 + c];
  }

  const int lane = t & 31, w = t >> 5;  // w = head
  const int l15 = lane & 15, hb = lane >> 4;

  v8f pacc[8];
  #pragma unroll
  for (int nt = 0; nt < 8; ++nt) pacc[nt] = (v8f){};

  for (int s = 0; s < kS; ++s) {
    __syncthreads();  // previous tile fully consumed (also covers attLds)
    if (t < 32) {
      tdm_load_xact_tile(Xact + ((size_t)b0 * kS + s) * kHID,
                         (unsigned)(uintptr_t)&xLds[0]);
      __builtin_amdgcn_s_wait_tensorcnt(0);
    }
    __syncthreads();  // tile visible to all waves

    float a[8];
    #pragma unroll
    for (int i = 0; i < 8; ++i) a[i] = attLds[(i + hb * 8) * 64 + s * 8 + w];

    const char* rowb = (const char*)&xLds[0] + l15 * kRowB;
    for (int nt = 0; nt < 8; ++nt) {
      const int n = w * 128 + nt * 16 + l15;
      const __bf16* bp = Wvt + ((size_t)s * kHOE + n) * kHID + hb * 16;
      __builtin_prefetch(bp + 16 * kHID, 0, 1);  // next nt's B tile
      v8f tmp = {};
      #pragma unroll
      for (int ks = 0; ks < 32; ++ks) {
        BF16x16 Av = lds_a_operand(rowb, ks, hb);
        BF16x16 Bv;
        Bv.u[0] = *(const v4u*)(bp + ks * 32);
        Bv.u[1] = *(const v4u*)(bp + ks * 32 + 8);
        tmp = wmma_bf16(Av.v, Bv.v, tmp);
      }
      const float vb = bv[s * kHOE + n];
      #pragma unroll
      for (int i = 0; i < 8; ++i) pacc[nt][i] += a[i] * (tmp[i] + vb);
    }
  }

  #pragma unroll
  for (int nt = 0; nt < 8; ++nt) {
    const int n = w * 128 + nt * 16 + l15;
    #pragma unroll
    for (int i = 0; i < 8; ++i) {
      int row = i + hb * 8;
      out[(size_t)(b0 + row) * kOROW + n] = pacc[nt][i];
    }
  }
}

// ---------------------------------------------------------------------------
// Host-side launcher. Workspace layout (assumes ws_size >= ~168 MB):
//   xact bf16 (128MB) | scores f32 (2MB) | att f32 (2MB) |
//   WtSub bf16 (4MB)  | Wqt (8MB) | Wkt (8MB) | Wvt (16MB)
// ---------------------------------------------------------------------------
extern "C" void kernel_launch(void* const* d_in, const int* in_sizes, int n_in,
                              void* d_out, int out_size, void* d_ws, size_t ws_size,
                              hipStream_t stream) {
  const float* x     = (const float*)d_in[0];
  const float* W_sub = (const float*)d_in[1];
  const float* b_sub = (const float*)d_in[2];
  const float* Wq    = (const float*)d_in[3];
  const float* bq    = (const float*)d_in[4];
  const float* Wk    = (const float*)d_in[5];
  const float* bk    = (const float*)d_in[6];
  const float* Wv    = (const float*)d_in[7];
  const float* bv    = (const float*)d_in[8];
  float* out = (float*)d_out;

  char* ws = (char*)d_ws;
  size_t off = 0;
  __bf16* Xact   = (__bf16*)(ws + off); off += (size_t)kB * kS * kHID * 2;
  float*  Scores = (float*) (ws + off); off += (size_t)kB * kS * kH * 4;
  float*  Att    = (float*) (ws + off); off += (size_t)kB * kS * kH * 4;
  __bf16* WtSub  = (__bf16*)(ws + off); off += (size_t)kS * kHID * kIN * 2;
  __bf16* Wqt    = (__bf16*)(ws + off); off += (size_t)kS * kHAE * kHID * 2;
  __bf16* Wkt    = (__bf16*)(ws + off); off += (size_t)kS * kHAE * kHID * 2;
  __bf16* Wvt    = (__bf16*)(ws + off); off += (size_t)kS * kHOE * kHID * 2;

  long tot;
  tot = (long)kS * kHID * kIN;
  transpose_bf16_kernel<<<(tot + 255) / 256, 256, 0, stream>>>(W_sub, WtSub, kIN, kHID, tot);
  tot = (long)kS * kHAE * kHID;
  transpose_bf16_kernel<<<(tot + 255) / 256, 256, 0, stream>>>(Wq, Wqt, kHID, kHAE, tot);
  transpose_bf16_kernel<<<(tot + 255) / 256, 256, 0, stream>>>(Wk, Wkt, kHID, kHAE, tot);
  tot = (long)kS * kHOE * kHID;
  transpose_bf16_kernel<<<(tot + 255) / 256, 256, 0, stream>>>(Wv, Wvt, kHID, kHOE, tot);

  sub_gemm_kernel<<<dim3(kS, kB / 64), 256, 0, stream>>>(x, WtSub, b_sub, Xact);
  qk_score_kernel<<<dim3(kS, kB / 16), 256, 0, stream>>>(Xact, Wqt, bq, Wkt, bk, Scores);
  att_kernel<<<(kB * kH) / 256, 256, 0, stream>>>(x, Scores, Att, out);
  pool_kernel<<<kB / 16, 256, 0, stream>>>(Xact, Wvt, bv, Att, out);
}
